// SwinBlock3D_89034672046687
// MI455X (gfx1250) — matrix operand order repaired and checked by reading, not compiled, not run on hardware
//
#include <hip/hip_runtime.h>
#include <hip/hip_bf16.h>

typedef __attribute__((ext_vector_type(16))) __bf16 v16bf;
typedef __attribute__((ext_vector_type(8)))  float  v8f;
typedef __attribute__((ext_vector_type(4)))  unsigned int v4u;
typedef __attribute__((ext_vector_type(8)))  int v8i;
typedef __attribute__((ext_vector_type(4)))  int v4i;
typedef unsigned int   u32;
typedef unsigned short u16;

#define WMMA_BF16(a, b, c) \
  __builtin_amdgcn_wmma_f32_16x16x32_bf16(false, (a), false, (b), (short)0, (c), false, false)

__device__ __forceinline__ u16 f2bf(float f) {
  u32 u = __float_as_uint(f);
  u32 r = (u + 0x7FFFu + ((u >> 16) & 1u)) >> 16;   // round-to-nearest-even
  return (u16)r;
}

// ---- Tensor Data Mover: 2D tile (tile_m rows x tile_k bf16) global->LDS.
// Pads LDS rows: every 32 dwords (+64 B16) insert 4 dwords -> LDS row stride 72 B16.
// D# packing per CDNA5 ISA 8.3/8.4 (type=2, data_size=1(2B), pad_en, interval=4(32dw), amount=3(4dw)).
__device__ __forceinline__ void tdm_load_2d(u32 lds_off, const u16* g,
                                            u32 tile_k, u32 tile_m, u32 stride_elems) {
  unsigned long long ga = (unsigned long long)(size_t)g;
  v4u g0;
  g0.x = 1u;                                              // count=1 (valid user D#)
  g0.y = lds_off;                                         // LDS byte address
  g0.z = (u32)ga;                                         // global_addr[31:0]
  g0.w = ((u32)(ga >> 32) & 0x01FFFFFFu) | (2u << 30);    // global_addr[56:32] | type=2
  v8i g1;
  g1[0] = (int)((1u << 16) | (1u << 20) | (4u << 22) | (3u << 25));
  g1[1] = (int)((tile_k & 0xFFFFu) << 16);                                   // tensor_dim0 lo
  g1[2] = (int)(((tile_k >> 16) & 0xFFFFu) | ((tile_m & 0xFFFFu) << 16));    // td0 hi | td1 lo
  g1[3] = (int)(((tile_m >> 16) & 0xFFFFu) | ((tile_k & 0xFFFFu) << 16));    // td1 hi | tile_dim0
  g1[4] = (int)(tile_m & 0xFFFFu);                                           // tile_dim1 (dim2=0)
  g1[5] = (int)stride_elems;                                                 // tensor_dim0_stride
  g1[6] = 0;
  g1[7] = 0;
  v4i z4 = {0, 0, 0, 0};
#if __clang_major__ >= 23
  v8i z8 = {0, 0, 0, 0, 0, 0, 0, 0};
  __builtin_amdgcn_tensor_load_to_lds(g0, g1, z4, z4, z8, 0);
#else
  __builtin_amdgcn_tensor_load_to_lds(g0, g1, z4, z4, 0);
#endif
}

// A-fragment (16x32 bf16, MxK), k = 16*half + s, from row-major LDS tile (row stride rs B16)
__device__ __forceinline__ v16bf load_a_frag(const u16* base, int rs) {
  int lane = threadIdx.x & 31;
  int m = lane & 15, hf = lane >> 4;
  union { v16bf v; u32 u[8]; } f;
  const u16* row = base + m * rs;
#pragma unroll
  for (int r = 0; r < 8; ++r) {
    int k = hf * 16 + 2 * r;
    f.u[r] = *(const u32*)(row + k);
  }
  return f.v;
}

// B-fragment (32x16, KxN) where the SOURCE is row-major with N as rows (i.e. B^T storage):
// K-pairs are contiguous.  base points at element [row n0][col kk0].
__device__ __forceinline__ v16bf load_b_fragT(const u16* base, int rs) {
  int lane = threadIdx.x & 31;
  int n = lane & 15, hf = lane >> 4;
  union { v16bf v; u32 u[8]; } f;
  const u16* row = base + n * rs;
#pragma unroll
  for (int r = 0; r < 8; ++r) {
    int kk = hf * 16 + 2 * r;
    f.u[r] = *(const u32*)(row + kk);
  }
  return f.v;
}

// B-fragment from pair-interleaved dwords bp[kpair*stride + n] = {B[2kp][n], B[2kp+1][n]}
__device__ __forceinline__ v16bf load_b_frag(const u32* bp, int stride, int n0, int kp0) {
  int lane = threadIdx.x & 31;
  int n = lane & 15, hf = lane >> 4;
  union { v16bf v; u32 u[8]; } f;
#pragma unroll
  for (int r = 0; r < 8; ++r)
    f.u[r] = bp[(kp0 + hf * 8 + r) * stride + n0 + n];
  return f.v;
}

// ---------------- weight prep: fp32 -> bf16 TRANSPOSED ([out_ch][in_ch]) ----------------
__global__ __launch_bounds__(256) void k_prep(
    const float* __restrict__ Wq, const float* __restrict__ Wk, const float* __restrict__ Wv,
    const float* __restrict__ Wo, const float* __restrict__ W1, const float* __restrict__ W2,
    const float* __restrict__ bq, const float* __restrict__ bk, const float* __restrict__ bv,
    u16* __restrict__ wqkvT, u16* __restrict__ woT, u16* __restrict__ w1T, u16* __restrict__ w2T,
    float* __restrict__ bqkv) {
  long i = (long)blockIdx.x * 256 + threadIdx.x;
  const long NQKV = 1152L * 384, NO = 384L * 384, N1 = 1536L * 384, N2 = 384L * 1536;
  if (i < NQKV) {                       // wqkvT[j][k]
    long j = i / 384, k = i % 384;
    float v = (j < 384) ? Wq[k * 384 + j] : (j < 768) ? Wk[k * 384 + (j - 384)] : Wv[k * 384 + (j - 768)];
    wqkvT[i] = f2bf(v); return;
  }
  i -= NQKV;
  if (i < NO) { long j = i / 384, k = i % 384; woT[i] = f2bf(Wo[k * 384 + j]); return; }
  i -= NO;
  if (i < N1) { long j = i / 384, k = i % 384; w1T[i] = f2bf(W1[k * 1536 + j]); return; }
  i -= N1;
  if (i < N2) { long j = i / 1536, k = i % 1536; w2T[i] = f2bf(W2[k * 384 + j]); return; }
  i -= N2;
  if (i < 1152) bqkv[i] = (i < 384) ? bq[i] : (i < 768) ? bk[i - 384] : bv[i - 768];
}

// ---------------- cyclic shift + (un)window ----------------
__global__ __launch_bounds__(256) void k_shift(const float* __restrict__ src,
                                               float* __restrict__ dst, int dir) {
  long idx = (long)blockIdx.x * 256 + threadIdx.x;     // over 384*16*64*64
  int c = (int)(idx >> 16);
  int rem = (int)(idx & 65535);
  int d = rem >> 12, h = (rem >> 6) & 63, w = rem & 63;
  int dd = (d + 14) & 15;
  int hh = (h + 60) & 63;
  int ww = (w + 60) & 63;
  int n = (((dd >> 2) * 8 + (hh >> 3)) * 8) + (ww >> 3);
  int t = (((dd & 3) * 8 + (hh & 7)) * 8) + (ww & 7);
  size_t xi = ((size_t)n * 256 + t) * 384 + c;
  if (dir == 0) dst[xi] = src[idx];
  else          dst[idx] = src[xi];
}

// ---------------- LayerNorm: one wave per token ----------------
// Output row staged in LDS, written out with async LDS->global burst stores.
__global__ __launch_bounds__(256) void k_ln(const float* __restrict__ xw,
                                            const float* __restrict__ g,
                                            const float* __restrict__ b,
                                            u16* __restrict__ hbf) {
  __shared__ __align__(16) u16 lbuf[8 * 384];
  int wv = threadIdx.x >> 5, lane = threadIdx.x & 31;
  size_t row = (size_t)blockIdx.x * 8 + wv;
  const float* x = xw + row * 384;
  float s = 0.f, s2 = 0.f, v[12];
#pragma unroll
  for (int i = 0; i < 12; ++i) { float t = x[i * 32 + lane]; v[i] = t; s += t; s2 += t * t; }
#pragma unroll
  for (int m = 1; m < 32; m <<= 1) { s += __shfl_xor(s, m, 32); s2 += __shfl_xor(s2, m, 32); }
  float mu = s * (1.0f / 384.0f);
  float var = s2 * (1.0f / 384.0f) - mu * mu;
  float rstd = rsqrtf(var + 1e-5f);
  u16* lb = lbuf + wv * 384;
#pragma unroll
  for (int i = 0; i < 12; ++i) {
    int c = i * 32 + lane;
    lb[c] = f2bf((v[i] - mu) * rstd * g[c] + b[c]);
  }
  asm volatile("s_wait_dscnt 0" ::: "memory");
  {
    u32 loff = (u32)(size_t)&lb[lane * 8];
    const u16* gp = hbf + row * 384 + lane * 8;
    asm volatile("global_store_async_from_lds_b128 %0, %1, off" :: "v"(gp), "v"(loff) : "memory");
    if (lane < 16) {
      u32 loff2 = (u32)(size_t)&lb[256 + lane * 8];
      const u16* gp2 = hbf + row * 384 + 256 + lane * 8;
      asm volatile("global_store_async_from_lds_b128 %0, %1, off" :: "v"(gp2), "v"(loff2) : "memory");
    }
  }
  asm volatile("s_wait_asynccnt 0" ::: "memory");
}

// ---------------- WMMA GEMM with double-buffered TDM tile loads ----------------
// C[M,N] = A[M,K] * W[K,N] + bias, where W is given TRANSPOSED: BT[N][K].
// EPI 0: bf16 out   EPI 1: bf16 + exact GELU   EPI 2: fp32 residual accumulate (ld 384)
template <int EPI>
__global__ __launch_bounds__(256) void k_gemm(const u16* __restrict__ A, const u16* __restrict__ BT,
                                              const float* __restrict__ bias, void* __restrict__ outp,
                                              int N, int K) {
  __shared__ __align__(16) u16 As[2][128 * 72];   // 128 x 64 bf16, TDM-padded stride 72
  __shared__ __align__(16) u16 Bs[2][64 * 72];    // 64(N) x 64(K) bf16 of BT
  int tid = threadIdx.x;
  int wv = tid >> 5;
  int m0 = blockIdx.x * 128;
  int n0 = blockIdx.y * 64;
  v8f z = {0.f, 0.f, 0.f, 0.f, 0.f, 0.f, 0.f, 0.f};
  v8f acc[4] = {z, z, z, z};

  // prologue: fill buffer 0
  if (wv == 0)      tdm_load_2d((u32)(size_t)&As[0][0], A + (size_t)m0 * K, 64, 128, (u32)K);
  else if (wv == 1) tdm_load_2d((u32)(size_t)&Bs[0][0], BT + (size_t)n0 * K, 64, 64, (u32)K);
  if (wv < 2) __builtin_amdgcn_s_wait_tensorcnt(0);
  __syncthreads();

  int cur = 0;
  for (int k0 = 0; k0 < K; k0 += 64) {
    if (k0 + 64 < K) {   // prefetch next tiles into the other buffer
      if (wv == 0)      tdm_load_2d((u32)(size_t)&As[cur ^ 1][0], A + (size_t)m0 * K + k0 + 64, 64, 128, (u32)K);
      else if (wv == 1) tdm_load_2d((u32)(size_t)&Bs[cur ^ 1][0], BT + (size_t)n0 * K + k0 + 64, 64, 64, (u32)K);
    }
#pragma unroll
    for (int kh = 0; kh < 2; ++kh) {
      v16bf a = load_a_frag(&As[cur][0] + wv * 16 * 72 + kh * 32, 72);
#pragma unroll
      for (int j = 0; j < 4; ++j) {
        v16bf b = load_b_fragT(&Bs[cur][0] + (j * 16) * 72 + kh * 32, 72);
        acc[j] = WMMA_BF16(a, b, acc[j]);
      }
    }
    if (wv < 2) __builtin_amdgcn_s_wait_tensorcnt(0);
    __syncthreads();
    cur ^= 1;
  }

  int lane = tid & 31, hf = lane >> 4, nn = lane & 15;
#pragma unroll
  for (int j = 0; j < 4; ++j)
#pragma unroll
    for (int r = 0; r < 8; ++r) {
      int gm = m0 + wv * 16 + hf * 8 + r;
      int gn = n0 + j * 16 + nn;
      float v = acc[j][r] + bias[gn];
      if (EPI == 1) v = 0.5f * v * (1.0f + erff(v * 0.70710678118f));
      if (EPI <= 1) ((u16*)outp)[(size_t)gm * N + gn] = f2bf(v);
      else ((float*)outp)[(size_t)gm * 384 + gn] += v;
    }
}

// ---------------- fused attention per (window, head) ----------------
// NOTE: deliberately no inline asm here — it forces the allocator off the
// >256-VGPR (MSB) path and spills the 16 score accumulators (seen in round 2).
__global__ __launch_bounds__(256) void k_attn(const u16* __restrict__ qkv, u16* __restrict__ ob) {
  __shared__ __align__(16) u16 qs[256 * 72];       // Q tile via TDM (row-major, stride 72)
  __shared__ __align__(16) u16 ks[256 * 72];       // K tile via TDM
  __shared__ __align__(16) u32 vsP[128 * 66];      // V pair-interleaved: [tokpair][c]
  __shared__ __align__(16) u16 psW[8 * 16 * 34];   // per-wave P chunk bounce buffer
  int tid = threadIdx.x;
  int h = blockIdx.x;                  // head 0..5
  int n = blockIdx.y;                  // window 0..255
  size_t rowbase = (size_t)n * 256 * 1152;
  size_t obase = (size_t)n * 256 * 384;
  int ho = h * 64;
  int wv = tid >> 5, lane = tid & 31, hf = lane >> 4, nn = lane & 15;

  // Q and K tiles: two parallel TDM descriptors (256 tokens x 64 ch, row stride 1152)
  if (wv == 0)      tdm_load_2d((u32)(size_t)qs, qkv + rowbase + ho, 64, 256, 1152);
  else if (wv == 1) tdm_load_2d((u32)(size_t)ks, qkv + rowbase + 384 + ho, 64, 256, 1152);
  // V: pair-interleave across tokens (not TDM-expressible), all waves
  {
    int c = tid & 63, tb = tid >> 6;
#pragma unroll 4
    for (int it = 0; it < 32; ++it) {
      int tp = tb + it * 4;
      u32 lo = qkv[rowbase + (size_t)(2 * tp) * 1152 + 768 + ho + c];
      u32 hi = qkv[rowbase + (size_t)(2 * tp + 1) * 1152 + 768 + ho + c];
      vsP[tp * 66 + c] = lo | (hi << 16);
    }
  }
  if (wv < 2) __builtin_amdgcn_s_wait_tensorcnt(0);
  __syncthreads();

  u16* myps = psW + wv * (16 * 34);
  v8f z = {0.f, 0.f, 0.f, 0.f, 0.f, 0.f, 0.f, 0.f};

  for (int rt = 0; rt < 2; ++rt) {
    int m0 = (wv + rt * 8) * 16;
    v16bf aq0 = load_a_frag(qs + m0 * 72, 72);
    v16bf aq1 = load_a_frag(qs + m0 * 72 + 32, 72);

    v8f s[16];
#pragma unroll
    for (int j = 0; j < 16; ++j) s[j] = z;
#pragma unroll
    for (int j = 0; j < 16; ++j) {
      v16bf b0 = load_b_fragT(ks + (j * 16) * 72, 72);
      s[j] = WMMA_BF16(aq0, b0, s[j]);
      v16bf b1 = load_b_fragT(ks + (j * 16) * 72 + 32, 72);
      s[j] = WMMA_BF16(aq1, b1, s[j]);
    }

    // softmax (row lives in one 16-lane group of the WMMA C layout)
    float mx[8], sm[8];
#pragma unroll
    for (int r = 0; r < 8; ++r) mx[r] = -1e30f;
#pragma unroll
    for (int j = 0; j < 16; ++j)
#pragma unroll
      for (int r = 0; r < 8; ++r) {
        float v = s[j][r] * 0.125f;
        s[j][r] = v;
        mx[r] = fmaxf(mx[r], v);
      }
#pragma unroll
    for (int m = 1; m < 16; m <<= 1)
#pragma unroll
      for (int r = 0; r < 8; ++r) mx[r] = fmaxf(mx[r], __shfl_xor(mx[r], m, 32));
#pragma unroll
    for (int r = 0; r < 8; ++r) sm[r] = 0.f;
#pragma unroll
    for (int j = 0; j < 16; ++j)
#pragma unroll
      for (int r = 0; r < 8; ++r) {
        float e = __expf(s[j][r] - mx[r]);
        s[j][r] = e;
        sm[r] += e;
      }
#pragma unroll
    for (int m = 1; m < 16; m <<= 1)
#pragma unroll
      for (int r = 0; r < 8; ++r) sm[r] += __shfl_xor(sm[r], m, 32);
#pragma unroll
    for (int r = 0; r < 8; ++r) sm[r] = 1.0f / sm[r];

    // O = P @ V, streaming P through LDS in 32-token chunks
    v8f o[4] = {z, z, z, z};
    for (int kc = 0; kc < 8; ++kc) {
#pragma unroll
      for (int jj = 0; jj < 2; ++jj) {
        int j = kc * 2 + jj;
#pragma unroll
        for (int r = 0; r < 8; ++r)
          myps[(hf * 8 + r) * 34 + jj * 16 + nn] = f2bf(s[j][r] * sm[r]);
      }
      v16bf a = load_a_frag(myps, 34);
#pragma unroll
      for (int j = 0; j < 4; ++j) {
        v16bf b = load_b_frag(vsP, 66, j * 16, kc * 16);
        o[j] = WMMA_BF16(a, b, o[j]);
      }
    }

    // write O strip (bf16, coalesced 32B runs per accumulator register)
#pragma unroll
    for (int j = 0; j < 4; ++j)
#pragma unroll
      for (int r = 0; r < 8; ++r) {
        int gm = m0 + hf * 8 + r;
        ob[obase + (size_t)gm * 384 + ho + j * 16 + nn] = f2bf(o[j][r]);
      }
  }
}

extern "C" void kernel_launch(void* const* d_in, const int* in_sizes, int n_in,
                              void* d_out, int out_size, void* d_ws, size_t ws_size,
                              hipStream_t stream) {
  const float* x    = (const float*)d_in[0];
  const float* Wq   = (const float*)d_in[1];
  const float* bq   = (const float*)d_in[2];
  const float* Wk   = (const float*)d_in[3];
  const float* bk   = (const float*)d_in[4];
  const float* Wv   = (const float*)d_in[5];
  const float* bv   = (const float*)d_in[6];
  const float* Wo   = (const float*)d_in[7];
  const float* bo   = (const float*)d_in[8];
  const float* ln1g = (const float*)d_in[9];
  const float* ln1b = (const float*)d_in[10];
  const float* ln2g = (const float*)d_in[11];
  const float* ln2b = (const float*)d_in[12];
  const float* W1   = (const float*)d_in[13];
  const float* b1   = (const float*)d_in[14];
  const float* W2   = (const float*)d_in[15];
  const float* b2   = (const float*)d_in[16];
  float* out = (float*)d_out;

  char* ws = (char*)d_ws;
  float* xw    = (float*)(ws);                 // 65536x384 f32   (100,663,296 B)
  u16*   hbf   = (u16*)(ws + 100663296);       // 65536x384 bf16  ( 50,331,648 B)
  u16*   qkvb  = (u16*)(ws + 150994944);       // 65536x1152 bf16 (150,994,944 B)
  u16*   obf   = (u16*)(ws + 301989888);       // 65536x384 bf16  ( 50,331,648 B)
  u16*   mlp   = qkvb;                         // 65536x1536 bf16 reuses [qkv|obf]
  u16*   wqkvT = (u16*)(ws + 352321536);       // 1152x384 bf16 (transposed)
  float* bqkv  = (float*)(ws + 353206272);     // 1152 f32
  u16*   woT   = (u16*)(ws + 353210880);       // 384x384 bf16 (transposed)
  u16*   w1T   = (u16*)(ws + 353505792);       // 1536x384 bf16 (transposed)
  u16*   w2T   = (u16*)(ws + 354685440);       // 384x1536 bf16 (transposed)

  k_prep<<<6917, 256, 0, stream>>>(Wq, Wk, Wv, Wo, W1, W2, bq, bk, bv, wqkvT, woT, w1T, w2T, bqkv);
  k_shift<<<98304, 256, 0, stream>>>(x, xw, 0);
  k_ln<<<8192, 256, 0, stream>>>(xw, ln1g, ln1b, hbf);
  k_gemm<0><<<dim3(512, 18), 256, 0, stream>>>(hbf, wqkvT, bqkv, (void*)qkvb, 1152, 384);
  k_attn<<<dim3(6, 256), 256, 0, stream>>>(qkvb, obf);
  k_gemm<2><<<dim3(512, 6), 256, 0, stream>>>(obf, woT, bo, (void*)xw, 384, 384);
  k_ln<<<8192, 256, 0, stream>>>(xw, ln2g, ln2b, hbf);
  k_gemm<1><<<dim3(512, 24), 256, 0, stream>>>(hbf, w1T, b1, (void*)mlp, 1536, 384);
  k_gemm<2><<<dim3(512, 6), 256, 0, stream>>>(mlp, w2T, b2, (void*)xw, 384, 1536);
  k_shift<<<98304, 256, 0, stream>>>(xw, out, 1);
}
